// SlidingWindowAttention_63926293233689
// MI455X (gfx1250) — matrix operand order repaired and checked by reading, not compiled, not run on hardware
//
#include <hip/hip_runtime.h>
#include <hip/hip_bf16.h>

// ---- CDNA5 (gfx1250) sliding-window attention, bf16 WMMA, fp32 softmax ----
// Double-buffered LDS staging + register prefetch: global-load waits land
// after the WMMA block, one workgroup barrier per 32-key chunk.

typedef __bf16 bf16_t;
typedef __attribute__((ext_vector_type(16))) __bf16 v16bf;
typedef __attribute__((ext_vector_type(8)))  __bf16 v8bf;
typedef __attribute__((ext_vector_type(8)))  float  v8f;

#define WINDOW   256
#define SEQ      8192
#define NHEAD    16
#define DHEAD    128
#define QTILE_WG 64            // 4 waves x 16 query rows
#define NCHUNK   18            // (64 + 2*256) / 32 key chunks per block
#define KROW     136           // K chunk LDS row: 128 + 8 pad (272B -> conflict-free)
#define VROW     40            // V^T LDS row: 32 + 8 pad (80B -> conflict-free)
#define PROW     40            // P scratch row pad

__global__ __launch_bounds__(128)
void swa_wmma_kernel(const float* __restrict__ q,
                     const float* __restrict__ k,
                     const float* __restrict__ v,
                     float* __restrict__ out)
{
    __shared__ __align__(16) bf16_t Ks[2][32 * KROW];     // double-buffered K chunk
    __shared__ __align__(16) bf16_t VT[2][DHEAD * VROW];  // double-buffered V^T chunk
    __shared__ __align__(16) bf16_t PS[4 * 16 * PROW];    // per-wave P scratch

    const int tid  = threadIdx.x;
    const int lane = tid & 31;
    const int wv   = tid >> 5;

    const int nqb = SEQ / QTILE_WG;                 // 128 query blocks per head
    const int h   = blockIdx.x / nqb;
    const int q0  = (blockIdx.x % nqb) * QTILE_WG;
    const int kv0 = q0 - WINDOW;

    const size_t headOff = (size_t)h * SEQ * DHEAD;
    const float* qh = q   + headOff;
    const float* kh = k   + headOff;
    const float* vh = v   + headOff;
    float*       oh = out + headOff;

    const int mrow = lane & 15;        // A row / B col / C col for this lane
    const int hi   = lane >> 4;        // half-wave select
    const int kb8  = hi * 8;           // A-fragment K base (16-bit A layout)
    const int kb16 = hi * 16;          // unused; B fragments also index with kb8

    // ---- Q A-fragments: 16x32 bf16 per WMMA, 4 chunks covering D=128
    const int qi_a = q0 + wv * 16 + mrow;
    v16bf aq[4];
    {
        const float* qrow = qh + (size_t)qi_a * DHEAD;
        #pragma unroll
        for (int c = 0; c < 4; ++c) {
            const float* b0 = qrow + 32 * c + kb8;
            const float* b1 = b0 + 16;
            #pragma unroll
            for (int i = 0; i < 8; ++i) {
                aq[c][i]     = (bf16_t)b0[i];
                aq[c][i + 8] = (bf16_t)b1[i];
            }
        }
    }

    // ---- accumulators
    v8f o[8];
    #pragma unroll
    for (int dt = 0; dt < 8; ++dt)
        #pragma unroll
        for (int i = 0; i < 8; ++i) o[dt][i] = 0.f;
    float rs[8];
    #pragma unroll
    for (int r = 0; r < 8; ++r) rs[r] = 0.f;

    // per-row lower window bound (hoisted out of the chunk loop)
    const int qbase = q0 + wv * 16;
    int qlo[8];
    #pragma unroll
    for (int r = 0; r < 8; ++r) qlo[r] = qbase + r + hi * 8 - WINDOW;

    // scale folded with log2(e): exp(s/sqrt(128)) == exp2(s * sc2)
    const float sc2 = (float)(0.08838834764831845 * 1.4426950408889634);

    // staging assignments
    const int sKey = tid >> 2;             // K staging: key row 0..31
    const int sSeg = (tid & 3) * 32;       // K staging: d segment
    const int vKey = tid & 31;             // V staging: key
    const int vSeg = (tid >> 5) * 32;      // V staging: d segment

    // ---- prefetch helpers (raw fp32 held in regs; cvt deferred to store step
    //      so the loadcnt wait lands after the WMMA block)
    auto loadChunk = [&](int ch, float4 kf[8], float4 vf[8]) {
        const int k0 = kv0 + ch * 32;
        const int kiK = k0 + sKey;
        if (kiK >= 0 && kiK < SEQ) {
            const float4* src = (const float4*)(kh + (size_t)kiK * DHEAD + sSeg);
            #pragma unroll
            for (int p = 0; p < 8; ++p) kf[p] = src[p];
        } else {
            #pragma unroll
            for (int p = 0; p < 8; ++p) kf[p] = make_float4(0.f, 0.f, 0.f, 0.f);
        }
        const int kiV = k0 + vKey;
        if (kiV >= 0 && kiV < SEQ) {
            const float4* src = (const float4*)(vh + (size_t)kiV * DHEAD + vSeg);
            #pragma unroll
            for (int p = 0; p < 8; ++p) vf[p] = src[p];
        } else {
            #pragma unroll
            for (int p = 0; p < 8; ++p) vf[p] = make_float4(0.f, 0.f, 0.f, 0.f);
        }
    };

    auto storeChunk = [&](int buf, const float4 kf[8], const float4 vf[8]) {
        // K: row-major bf16, 4x ds_store_b128 per lane
        bf16_t* kd = &Ks[buf][sKey * KROW + sSeg];
        #pragma unroll
        for (int p = 0; p < 4; ++p) {
            v8bf t;
            t[0] = (bf16_t)kf[2*p].x;   t[1] = (bf16_t)kf[2*p].y;
            t[2] = (bf16_t)kf[2*p].z;   t[3] = (bf16_t)kf[2*p].w;
            t[4] = (bf16_t)kf[2*p+1].x; t[5] = (bf16_t)kf[2*p+1].y;
            t[6] = (bf16_t)kf[2*p+1].z; t[7] = (bf16_t)kf[2*p+1].w;
            *(v8bf*)(kd + p * 8) = t;
        }
        // V: transposed scatter, b16 stores
        bf16_t* vd = &VT[buf][0];
        #pragma unroll
        for (int p = 0; p < 8; ++p) {
            vd[(vSeg + 4*p + 0) * VROW + vKey] = (bf16_t)vf[p].x;
            vd[(vSeg + 4*p + 1) * VROW + vKey] = (bf16_t)vf[p].y;
            vd[(vSeg + 4*p + 2) * VROW + vKey] = (bf16_t)vf[p].z;
            vd[(vSeg + 4*p + 3) * VROW + vKey] = (bf16_t)vf[p].w;
        }
    };

    // ---- prologue: stage chunk 0
    {
        float4 kf[8], vf[8];
        loadChunk(0, kf, vf);
        storeChunk(0, kf, vf);
    }

    for (int ch = 0; ch < NCHUNK; ++ch) {
        const int k0  = kv0 + ch * 32;
        const int buf = ch & 1;

        // 1) issue next chunk's global loads (uniform guard, EXEC stays full)
        float4 kf[8], vf[8];
        const bool more = (ch + 1) < NCHUNK;
        if (more) loadChunk(ch + 1, kf, vf);

        // 2) barrier: publishes buf[ch&1], retires reads of buf[(ch+1)&1]
        __syncthreads();

        // 3) S = Q K^T : two 16-key tiles, 4 WMMAs each over D=128
        v8f s[2];
        #pragma unroll
        for (int j = 0; j < 2; ++j) {
            #pragma unroll
            for (int i = 0; i < 8; ++i) s[j][i] = 0.f;
            #pragma unroll
            for (int c = 0; c < 4; ++c) {
                const bf16_t* kp = &Ks[buf][(j * 16 + mrow) * KROW + 32 * c + kb8];
                v8bf b0 = *(const v8bf*)kp;
                v8bf b1 = *(const v8bf*)(kp + 16);
                v16bf bb;
                *(v8bf*)&bb       = b0;
                *((v8bf*)&bb + 1) = b1;
                s[j] = __builtin_amdgcn_wmma_f32_16x16x32_bf16(
                    false, aq[c], false, bb, (short)0, s[j], false, false);
            }
        }

        // 4) masked exp2 (unsigned range trick: one sub+cmp per element)
        #pragma unroll
        for (int j = 0; j < 2; ++j) {
            const int  ki   = k0 + j * 16 + mrow;
            const bool kiOK = (unsigned)ki < (unsigned)SEQ;
            #pragma unroll
            for (int r = 0; r < 8; ++r) {
                const bool ok = kiOK & ((unsigned)(ki - qlo[r]) <= 2u * WINDOW);
                float e = __builtin_amdgcn_exp2f(s[j][r] * sc2);
                e = ok ? e : 0.f;
                rs[r] += e;
                PS[(wv * 16 + r + hi * 8) * PROW + j * 16 + mrow] = (bf16_t)e;
            }
        }
        // per-wave LDS is in-order; wait for stores before A-layout reload
        asm volatile("s_wait_dscnt 0x0" ::: "memory");

        // 5) reload P as A-fragment (C->A transpose via LDS)
        v16bf ap;
        {
            const bf16_t* pr = &PS[(wv * 16 + mrow) * PROW + kb8];
            v8bf p0 = *(const v8bf*)pr;
            v8bf p1 = *(const v8bf*)(pr + 16);
            *(v8bf*)&ap       = p0;
            *((v8bf*)&ap + 1) = p1;
        }

        // 6) O += P V : 8 output d-tiles, one K=32 WMMA each
        #pragma unroll
        for (int dt = 0; dt < 8; ++dt) {
            const bf16_t* vp = &VT[buf][(dt * 16 + mrow) * VROW + kb8];
            v8bf b0 = *(const v8bf*)vp;
            v8bf b1 = *(const v8bf*)(vp + 16);
            v16bf bb;
            *(v8bf*)&bb       = b0;
            *((v8bf*)&bb + 1) = b1;
            o[dt] = __builtin_amdgcn_wmma_f32_16x16x32_bf16(
                false, ap, false, bb, (short)0, o[dt], false, false);
        }

        // 7) convert + store next chunk into the alternate buffer
        //    (global loadcnt wait lands here, hidden behind the WMMAs above)
        if (more) storeChunk(buf ^ 1, kf, vf);
    }

    // ---- normalize rows and store fp32 output
    #pragma unroll
    for (int r = 0; r < 8; ++r) {
        float t = rs[r];
        t += __shfl_xor(t, 1, 32);
        t += __shfl_xor(t, 2, 32);
        t += __shfl_xor(t, 4, 32);
        t += __shfl_xor(t, 8, 32);
        const float inv = 1.0f / t;
        const int qi = q0 + wv * 16 + r + hi * 8;
        float* orow = oh + (size_t)qi * DHEAD + mrow;
        #pragma unroll
        for (int dt = 0; dt < 8; ++dt)
            orow[dt * 16] = o[dt][r] * inv;
    }
}

extern "C" void kernel_launch(void* const* d_in, const int* in_sizes, int n_in,
                              void* d_out, int out_size, void* d_ws, size_t ws_size,
                              hipStream_t stream) {
    const float* q = (const float*)d_in[0];
    const float* k = (const float*)d_in[1];
    const float* v = (const float*)d_in[2];
    float* out = (float*)d_out;

    const int blocks = NHEAD * (SEQ / QTILE_WG);   // 16 * 128 = 2048
    swa_wmma_kernel<<<blocks, 128, 0, stream>>>(q, k, v, out);
}
